// swinU3_61701500174703
// MI455X (gfx1250) — matrix operand order repaired and checked by reading, not compiled, not run on hardware
//
#include <hip/hip_runtime.h>
#include <cstdint>
#include <cstddef>

typedef __bf16 bf16;
typedef __attribute__((ext_vector_type(16))) __bf16 v16bf;
typedef __attribute__((ext_vector_type(8)))  __bf16 v8bf;
typedef __attribute__((ext_vector_type(8)))  float  v8f;
typedef __attribute__((ext_vector_type(4)))  unsigned int v4u;
typedef __attribute__((ext_vector_type(8)))  int v8i;
typedef __attribute__((ext_vector_type(4)))  int v4i;

#define L_TOK 32768
#define CDIM  384
#define NHEAD 8
#define HD    48
#define HDP   64
#define NWIN  64
#define NTOK  512
#define CH3   1152
#define CMLP  1536

#define EP_BF16   0
#define EP_GELU   1
#define EP_F32RES 2

__device__ __forceinline__ float bf2f(bf16 v){ return (float)v; }
__device__ __forceinline__ bf16  f2bf(float f){ return (bf16)f; }

// Concatenate two 8-element bf16 chunks into a WMMA 16-element fragment register pair.
__device__ __forceinline__ v16bf cat16(v8bf lo, v8bf hi) {
  return __builtin_shufflevector(lo, hi, 0,1,2,3,4,5,6,7,8,9,10,11,12,13,14,15);
}

// ---------------------------------------------------------------- Tensor Data Mover (ISA ch.8)
// 2D tile DMA global -> LDS, bf16 elements (data_size code 1 = 2 bytes).
// dim0/stride0 along the contiguous axis; tile laid out densely in LDS.
// 6-arg builtin form: (g0, g1, g2, g3, g4, cpol).
__device__ __forceinline__ void tdm_load_2d(
    unsigned lds_off, const void* gaddr,
    unsigned dim0, unsigned dim1, unsigned stride0,
    unsigned tile0, unsigned tile1)
{
  const unsigned long long ga = (unsigned long long)(uintptr_t)gaddr;
  v4u g0;
  g0[0] = 1u;                                           // count=1, user-mode, no gather
  g0[1] = lds_off;                                      // LDS byte address
  g0[2] = (unsigned)(ga & 0xFFFFFFFFu);                 // global addr [31:0]
  g0[3] = (unsigned)((ga >> 32) & 0x1FFFFFFu) | (2u << 30);  // addr[56:32] | type=2
  v8i g1;
  g1[0] = (int)(1u << 16);                              // wg_mask=0, data_size=1 (2B)
  g1[1] = (int)((dim0 & 0xFFFFu) << 16);                // tensor_dim0 lo16 (bits 63:48)
  g1[2] = (int)(((dim0 >> 16) & 0xFFFFu) | ((dim1 & 0xFFFFu) << 16)); // dim0 hi | dim1 lo
  g1[3] = (int)(((dim1 >> 16) & 0xFFFFu) | ((tile0 & 0xFFFFu) << 16)); // dim1 hi | tile0
  g1[4] = (int)(tile1 & 0xFFFFu);                       // tile_dim1 | tile_dim2=0
  g1[5] = (int)stride0;                                 // tensor_dim0_stride lo32
  g1[6] = 0;                                            // stride0 hi16 | dim1_stride lo16
  g1[7] = 0;
  const v4i z4 = {0, 0, 0, 0};
  const v8i z8 = {0, 0, 0, 0, 0, 0, 0, 0};
  __builtin_amdgcn_tensor_load_to_lds(g0, g1, z4, z4, z8, 0);
}

// ---------------------------------------------------------------- LayerNorm
// One wave (32 lanes) per row of 384; 8 rows per 256-thread block.
__global__ __launch_bounds__(256) void ln_rows_bf16(
    const float* __restrict__ x, const float* __restrict__ g,
    const float* __restrict__ b, bf16* __restrict__ out)
{
  const int wv = threadIdx.x >> 5, lane = threadIdx.x & 31;
  const int row = blockIdx.x * 8 + wv;
  const float* xr = x + (size_t)row * CDIM;
  float s = 0.f, ss = 0.f, v[12];
#pragma unroll
  for (int i = 0; i < 12; i++) { v[i] = xr[lane + i*32]; s += v[i]; ss += v[i]*v[i]; }
#pragma unroll
  for (int o = 16; o >= 1; o >>= 1) { s += __shfl_xor(s, o, 32); ss += __shfl_xor(ss, o, 32); }
  const float mu = s * (1.f / CDIM);
  const float var = ss * (1.f / CDIM) - mu * mu;
  const float rstd = rsqrtf(var + 1e-5f);
  bf16* orow = out + (size_t)row * CDIM;
#pragma unroll
  for (int i = 0; i < 12; i++) {
    const int c = lane + i*32;
    orow[c] = f2bf((v[i] - mu) * rstd * g[c] + b[c]);
  }
}

// Final LN fused with [L,C] -> [C, H*W*T] transpose (fp32 output).
__global__ __launch_bounds__(256) void ln_final_tr(
    const float* __restrict__ x, const float* __restrict__ g,
    const float* __restrict__ b, float* __restrict__ out)
{
  const int wv = threadIdx.x >> 5, lane = threadIdx.x & 31;
  const int row = blockIdx.x * 8 + wv;
  const float* xr = x + (size_t)row * CDIM;
  float s = 0.f, ss = 0.f, v[12];
#pragma unroll
  for (int i = 0; i < 12; i++) { v[i] = xr[lane + i*32]; s += v[i]; ss += v[i]*v[i]; }
#pragma unroll
  for (int o = 16; o >= 1; o >>= 1) { s += __shfl_xor(s, o, 32); ss += __shfl_xor(ss, o, 32); }
  const float mu = s * (1.f / CDIM);
  const float var = ss * (1.f / CDIM) - mu * mu;
  const float rstd = rsqrtf(var + 1e-5f);
#pragma unroll
  for (int i = 0; i < 12; i++) {
    const int c = lane + i*32;
    out[(size_t)c * L_TOK + row] = (v[i] - mu) * rstd * g[c] + b[c];
  }
}

// Convert fp32 weight [K,N] -> bf16 transposed [N,K] (coalesced read, scattered write).
__global__ void cvt_tr_bf16(const float* __restrict__ in, bf16* __restrict__ out,
                            int K, int N)
{
  const int i = blockIdx.x * blockDim.x + threadIdx.x;
  if (i < K * N) {
    const int k = i / N, n = i - k * N;
    out[(size_t)n * K + k] = f2bf(in[i]);
  }
}

// ---------------------------------------------------------------- GEMM (bf16 x bf16^T -> f32 accum)
// A [M,K] row-major, Wt [N,K] row-major (pre-transposed weights).
// Block tile 128x128, 8 waves each computing 32x64 via v_wmma_f32_16x16x32_bf16.
// Tiles staged into LDS by the Tensor Data Mover (wave 0 issues, TENSORcnt waits).
__global__ __launch_bounds__(256) void gemm_bf16_wmma(
    const bf16* __restrict__ A, const bf16* __restrict__ Wt,
    const float* __restrict__ bias, const float* __restrict__ res,
    void* __restrict__ outp, int M, int N, int K, int mode)
{
  __shared__ __align__(32) char smem[16384];     // As 8KB @0, BsT 8KB @8192
  bf16 (*As)[32]  = (bf16(*)[32])(smem);
  bf16 (*BsT)[32] = (bf16(*)[32])(smem + 8192);
  const int tid = threadIdx.x;
  const int bm = blockIdx.y * 128, bn = blockIdx.x * 128;
  const int wv = tid >> 5, lane = tid & 31;
  const int hf = lane >> 4, l15 = lane & 15;
  const int wm = (wv >> 1) * 32, wn = (wv & 1) * 64;

  v8f acc[2][4];
#pragma unroll
  for (int mt = 0; mt < 2; mt++)
#pragma unroll
    for (int nt = 0; nt < 4; nt++)
#pragma unroll
      for (int r = 0; r < 8; r++) acc[mt][nt][r] = 0.f;

  for (int k0 = 0; k0 < K; k0 += 32) {
    if (wv == 0) {
      // A tile: 128 rows x 32 k, row stride K elems.   B tile: same from Wt.
      tdm_load_2d(0,    A  + (size_t)bm * K + k0, 32, 128, (unsigned)K, 32, 128);
      tdm_load_2d(8192, Wt + (size_t)bn * K + k0, 32, 128, (unsigned)K, 32, 128);
      __builtin_amdgcn_s_wait_tensorcnt(0);
    }
    __syncthreads();
    // TDM writes LDS behind the compiler's back: escape smem + memory clobber so
    // the fragment loads below are materialized (not folded to undef).
    asm volatile("" :: "r"(smem) : "memory");

    // A fragment: 16x32 bf16 (ISA 7.12.2): two contiguous 8-elem runs per lane.
    v16bf afrag[2];
#pragma unroll
    for (int mt = 0; mt < 2; mt++) {
      const bf16* Ar = &As[wm + mt*16 + l15][0];
      const int kb = hf * 8;
      afrag[mt] = cat16(*(const v8bf*)&Ar[kb], *(const v8bf*)&Ar[16 + kb]);
    }
#pragma unroll
    for (int nt = 0; nt < 4; nt++) {
      // B fragment 32x16: two 8-elem k runs per lane (same order as A).
      const v16bf bfrag = cat16(*(const v8bf*)&BsT[wn + nt*16 + l15][hf * 8], *(const v8bf*)&BsT[wn + nt*16 + l15][16 + hf * 8]);
#pragma unroll
      for (int mt = 0; mt < 2; mt++)
        acc[mt][nt] = __builtin_amdgcn_wmma_f32_16x16x32_bf16(
            false, afrag[mt], false, bfrag, (short)0, acc[mt][nt], false, false);
    }
    __syncthreads();
  }

#pragma unroll
  for (int mt = 0; mt < 2; mt++)
#pragma unroll
    for (int nt = 0; nt < 4; nt++)
#pragma unroll
      for (int r = 0; r < 8; r++) {
        const int row = bm + wm + mt*16 + r + 8*hf;
        const int col = bn + wn + nt*16 + l15;
        float vo = acc[mt][nt][r] + bias[col];
        if (mode == EP_GELU) vo = 0.5f * vo * (1.f + erff(vo * 0.70710678118f));
        if (mode == EP_F32RES)
          ((float*)outp)[(size_t)row * N + col] = vo + res[(size_t)row * N + col];
        else
          ((bf16*)outp)[(size_t)row * N + col] = f2bf(vo);
      }
}

// ---------------------------------------------------------------- QKV gather (window partition + shift roll)
// Q/K layout: ((w*8+h)*512 + n)*64 + d.  V layout (transposed): (w*8+h)*32768 + d*512 + n.
__global__ __launch_bounds__(256) void gather_qkv(
    const bf16* __restrict__ qkv, bf16* __restrict__ q,
    bf16* __restrict__ k, bf16* __restrict__ v, int shift)
{
  const int gid = blockIdx.x * 256 + threadIdx.x;
  const int d = gid & 63, n = (gid >> 6) & 511, h = (gid >> 15) & 7, w = gid >> 18;
  const int wi = w >> 4, wj = (w >> 2) & 3, wk = w & 3;
  const int i = n >> 6, j = (n >> 3) & 7, kk = n & 7;
  int gh = wi*8 + i, gw = wj*8 + j, gt = wk*8 + kk;
  if (shift) { gh = (gh + 4) & 31; gw = (gw + 4) & 31; gt = (gt + 4) & 31; }
  const size_t l = (size_t)gh * 1024 + gw * 32 + gt;
  float qv = 0.f, kv = 0.f, vv = 0.f;
  if (d < HD) {
    const int c = h * HD + d;
    const bf16* base = qkv + l * CH3;
    qv = bf2f(base[c]) * 0.14433756729740643f;   // 48^-0.5 folded into q
    kv = bf2f(base[384 + c]);
    vv = bf2f(base[768 + c]);
  }
  q[gid] = f2bf(qv);
  k[gid] = f2bf(kv);
  const size_t whbase = (size_t)((w << 3) | h) * NTOK * HDP;
  v[whbase + (size_t)d * NTOK + n] = f2bf(vv);
}

// ---------------------------------------------------------------- rel-pos bias table [h][n][m]
__global__ __launch_bounds__(256) void bias_prep(
    const float* __restrict__ rpb, float* __restrict__ biasT)
{
  const int gid = blockIdx.x * 256 + threadIdx.x;       // (h*512+n)*512+m
  const int m = gid & 511, n = (gid >> 9) & 511, h = gid >> 18;
  const int ni = n >> 6, nj = (n >> 3) & 7, nk = n & 7;
  const int mi = m >> 6, mj = (m >> 3) & 7, mk = m & 7;
  const int idx = (ni - mi + 7) * 225 + (nj - mj + 7) * 15 + (nk - mk + 7);
  biasT[gid] = rpb[idx * 8 + h];
}

__device__ __forceinline__ int regc(int c) { return c < 24 ? 0 : (c < 28 ? 1 : 2); }

// ---------------------------------------------------------------- windowed attention (flash-style)
// One block per (window, head). 16 waves, each owning 32 query rows.
// K/V tiles DMA'd to LDS by the Tensor Data Mover.
__global__ __launch_bounds__(512) void attn_win(
    const bf16* __restrict__ Q, const bf16* __restrict__ K, const bf16* __restrict__ Vt,
    const float* __restrict__ biasT, bf16* __restrict__ ao, int use_mask, int shift)
{
  __shared__ __align__(32) char smem[49152];  // Ks 16KB @0, Vst 16KB @16384, Ps 16KB @32768
  bf16 (*Ks)[64]      = (bf16(*)[64])(smem);            // [key][d]
  bf16 (*Vst)[128]    = (bf16(*)[128])(smem + 16384);   // [d][key]
  bf16 (*Ps)[16][32]  = (bf16(*)[16][32])(smem + 32768);
  const int blk = blockIdx.x;
  const int w = blk >> 3, h = blk & 7;
  const int tid = threadIdx.x;
  const int wv = tid >> 5, lane = tid & 31;
  const int hf = lane >> 4, l15 = lane & 15;
  const size_t base = (size_t)blk * NTOK * HDP;
  const int wi = w >> 4, wj = (w >> 2) & 3, wk = w & 3;

  // Q fragments: rows wv*32 .. +31, d padded to 64 -> two k=32 frags per 16-row tile.
  v16bf aq[2][2];
#pragma unroll
  for (int rt = 0; rt < 2; rt++) {
    const bf16* Qr = Q + base + (size_t)(wv*32 + rt*16 + l15) * HDP;
    const int kb = hf * 8;
    aq[rt][0] = cat16(*(const v8bf*)&Qr[kb],      *(const v8bf*)&Qr[16 + kb]);
    aq[rt][1] = cat16(*(const v8bf*)&Qr[32 + kb], *(const v8bf*)&Qr[48 + kb]);
  }

  v8f acc[2][4];
  float mrow[2][8], lrow[2][8];
#pragma unroll
  for (int rt = 0; rt < 2; rt++)
#pragma unroll
    for (int r = 0; r < 8; r++) { mrow[rt][r] = -1e30f; lrow[rt][r] = 0.f; }
#pragma unroll
  for (int rt = 0; rt < 2; rt++)
#pragma unroll
    for (int nt = 0; nt < 4; nt++)
#pragma unroll
      for (int r = 0; r < 8; r++) acc[rt][nt][r] = 0.f;

  for (int jb = 0; jb < NTOK; jb += 128) {
    if (wv == 0) {
      // K tile: 128 keys x 64 d (stride 64).  V tile: 64 d x 128 keys (stride 512).
      tdm_load_2d(0,     K  + base + (size_t)jb * HDP, 64, 128, 64, 64, 128);
      tdm_load_2d(16384, Vt + base + jb, 128, 64, 512, 128, 64);
      __builtin_amdgcn_s_wait_tensorcnt(0);
    }
    __syncthreads();
    // Make the TDM's LDS writes visible to the optimizer (see gemm kernel).
    asm volatile("" :: "r"(smem) : "memory");

    for (int j2 = 0; j2 < 128; j2 += 32) {
#pragma unroll
      for (int rt = 0; rt < 2; rt++) {
        // ---- S = Q K^T (two 16-col tiles, each two k=32 WMMAs)
        v8f s[2];
#pragma unroll
        for (int t = 0; t < 2; t++) {
          const int krow = j2 + t*16 + l15;
          const v16bf bk0 = cat16(*(const v8bf*)&Ks[krow][hf * 8], *(const v8bf*)&Ks[krow][16 + hf * 8]);
          const v16bf bk1 = cat16(*(const v8bf*)&Ks[krow][32 + hf * 8], *(const v8bf*)&Ks[krow][48 + hf * 8]);
          v8f z;
#pragma unroll
          for (int r = 0; r < 8; r++) z[r] = 0.f;
          z    = __builtin_amdgcn_wmma_f32_16x16x32_bf16(false, aq[rt][0], false, bk0, (short)0, z, false, false);
          s[t] = __builtin_amdgcn_wmma_f32_16x16x32_bf16(false, aq[rt][1], false, bk1, (short)0, z, false, false);
        }
        // ---- bias + shift-window mask
        const int rowb = wv*32 + rt*16 + 8*hf;
#pragma unroll
        for (int t = 0; t < 2; t++) {
          const int mcol = jb + j2 + t*16 + l15;
          int rm = 0;
          if (use_mask) {
            const int ci = mcol >> 6, cj = (mcol >> 3) & 7, ck = mcol & 7;
            rm = regc(wi*8 + ci) * 9 + regc(wj*8 + cj) * 3 + regc(wk*8 + ck);
          }
#pragma unroll
          for (int r = 0; r < 8; r++) {
            const int nrow = rowb + r;
            float sv = s[t][r] + biasT[((size_t)h * NTOK + nrow) * NTOK + mcol];
            if (use_mask) {
              const int ri = nrow >> 6, rj = (nrow >> 3) & 7, rk = nrow & 7;
              const int rn = regc(wi*8 + ri) * 9 + regc(wj*8 + rj) * 3 + regc(wk*8 + rk);
              if (rn != rm) sv -= 100.f;
            }
            s[t][r] = sv;
          }
        }
        // ---- online softmax (row stats per lane-half; xor<=8 stays in half)
#pragma unroll
        for (int r = 0; r < 8; r++) {
          float mx = fmaxf(s[0][r], s[1][r]);
#pragma unroll
          for (int o = 8; o >= 1; o >>= 1) mx = fmaxf(mx, __shfl_xor(mx, o, 32));
          const float mnew = fmaxf(mrow[rt][r], mx);
          const float alpha = __expf(mrow[rt][r] - mnew);
          const float p0 = __expf(s[0][r] - mnew);
          const float p1 = __expf(s[1][r] - mnew);
          float ps = p0 + p1;
#pragma unroll
          for (int o = 8; o >= 1; o >>= 1) ps += __shfl_xor(ps, o, 32);
          lrow[rt][r] = lrow[rt][r] * alpha + ps;
          mrow[rt][r] = mnew;
#pragma unroll
          for (int nt = 0; nt < 4; nt++) acc[rt][nt][r] *= alpha;
          s[0][r] = p0; s[1][r] = p1;
        }
        // ---- transpose P (C layout -> A layout) through wave-private LDS
#pragma unroll
        for (int t = 0; t < 2; t++)
#pragma unroll
          for (int r = 0; r < 8; r++)
            Ps[wv][r + 8*hf][t*16 + l15] = f2bf(s[t][r]);
        v16bf pf;
        {
          const bf16* Pr = &Ps[wv][l15][0];
          const int kb = hf * 8;
          pf = cat16(*(const v8bf*)&Pr[kb], *(const v8bf*)&Pr[16 + kb]);
        }
        // ---- O += P V : B-frag two 8-key runs in Vst[d][key]
#pragma unroll
        for (int nt = 0; nt < 4; nt++) {
          const v16bf bv = cat16(*(const v8bf*)&Vst[nt*16 + l15][j2 + hf * 8], *(const v8bf*)&Vst[nt*16 + l15][j2 + 16 + hf * 8]);
          acc[rt][nt] = __builtin_amdgcn_wmma_f32_16x16x32_bf16(
              false, pf, false, bv, (short)0, acc[rt][nt], false, false);
        }
      }
    }
    __syncthreads();
  }

  // ---- normalize + scatter (window reverse + un-roll); proj is linear so scatter-then-proj is fine
#pragma unroll
  for (int rt = 0; rt < 2; rt++)
#pragma unroll
    for (int nt = 0; nt < 4; nt++)
#pragma unroll
      for (int r = 0; r < 8; r++) {
        const int d = nt*16 + l15;
        if (d >= HD) continue;
        const int nrow = wv*32 + rt*16 + r + 8*hf;
        const float ov = acc[rt][nt][r] / lrow[rt][r];
        const int i = nrow >> 6, j = (nrow >> 3) & 7, kk = nrow & 7;
        int gh = wi*8 + i, gw = wj*8 + j, gt = wk*8 + kk;
        if (shift) { gh = (gh + 4) & 31; gw = (gw + 4) & 31; gt = (gt + 4) & 31; }
        const size_t l = (size_t)gh * 1024 + gw * 32 + gt;
        ao[l * CDIM + h * HD + d] = f2bf(ov);
      }
}

// ---------------------------------------------------------------- host
extern "C" void kernel_launch(void* const* d_in, const int* in_sizes, int n_in,
                              void* d_out, int out_size, void* d_ws, size_t ws_size,
                              hipStream_t stream)
{
  (void)in_sizes; (void)n_in; (void)out_size; (void)ws_size;
  const float* x = (const float*)d_in[0];
  // params flattened: x, [w block: n1_g n1_b qkv_w qkv_b rpb proj_w proj_b n2_g n2_b fc1_w fc1_b fc2_w fc2_b],
  //                   [sw block same 13], norm_g, norm_b, H, W, T
  auto P = [&](int b, int off) -> const float* { return (const float*)d_in[1 + b*13 + off]; };
  const float* norm_g = (const float*)d_in[27];
  const float* norm_b = (const float*)d_in[28];

  char* p = (char*)d_ws;
  auto carve = [&](size_t bytes) -> char* {
    char* r = p; p += (bytes + 255) & ~(size_t)255; return r;
  };
  float* xA    = (float*)carve((size_t)L_TOK * CDIM * 4);
  float* xB    = (float*)carve((size_t)L_TOK * CDIM * 4);
  bf16*  xn    = (bf16*) carve((size_t)L_TOK * CDIM * 2);
  bf16*  qkv   = (bf16*) carve((size_t)L_TOK * CH3 * 2);
  bf16*  qb    = (bf16*) carve((size_t)NWIN * NHEAD * NTOK * HDP * 2);
  bf16*  kbuf  = (bf16*) carve((size_t)NWIN * NHEAD * NTOK * HDP * 2);
  bf16*  vbuf  = (bf16*) carve((size_t)NWIN * NHEAD * NTOK * HDP * 2);
  bf16*  ao    = (bf16*) carve((size_t)L_TOK * CDIM * 2);
  bf16*  hh    = (bf16*) carve((size_t)L_TOK * CMLP * 2);
  float* biasT = (float*)carve((size_t)NHEAD * NTOK * NTOK * 4);
  bf16*  wbf   = (bf16*) carve((size_t)CDIM * CMLP * 2);

  (void)hipMemcpyAsync(xA, x, (size_t)L_TOK * CDIM * 4, hipMemcpyDeviceToDevice, stream);

  for (int b = 0; b < 2; b++) {
    const int shift = b;
    ln_rows_bf16<<<L_TOK/8, 256, 0, stream>>>(xA, P(b,0), P(b,1), xn);
    cvt_tr_bf16<<<(CDIM*CH3 + 255)/256, 256, 0, stream>>>(P(b,2), wbf, CDIM, CH3);
    gemm_bf16_wmma<<<dim3(CH3/128, L_TOK/128), 256, 0, stream>>>(
        xn, wbf, P(b,3), nullptr, qkv, L_TOK, CH3, CDIM, EP_BF16);
    gather_qkv<<<(NWIN*NHEAD*NTOK*HDP)/256, 256, 0, stream>>>(qkv, qb, kbuf, vbuf, shift);
    bias_prep<<<(NHEAD*NTOK*NTOK)/256, 256, 0, stream>>>(P(b,4), biasT);
    attn_win<<<NWIN*NHEAD, 512, 0, stream>>>(qb, kbuf, vbuf, biasT, ao, shift, shift);
    cvt_tr_bf16<<<(CDIM*CDIM + 255)/256, 256, 0, stream>>>(P(b,5), wbf, CDIM, CDIM);
    gemm_bf16_wmma<<<dim3(CDIM/128, L_TOK/128), 256, 0, stream>>>(
        ao, wbf, P(b,6), xA, xB, L_TOK, CDIM, CDIM, EP_F32RES);
    ln_rows_bf16<<<L_TOK/8, 256, 0, stream>>>(xB, P(b,7), P(b,8), xn);
    cvt_tr_bf16<<<(CDIM*CMLP + 255)/256, 256, 0, stream>>>(P(b,9), wbf, CDIM, CMLP);
    gemm_bf16_wmma<<<dim3(CMLP/128, L_TOK/128), 256, 0, stream>>>(
        xn, wbf, P(b,10), nullptr, hh, L_TOK, CMLP, CDIM, EP_GELU);
    cvt_tr_bf16<<<(CMLP*CDIM + 255)/256, 256, 0, stream>>>(P(b,11), wbf, CMLP, CDIM);
    gemm_bf16_wmma<<<dim3(CDIM/128, L_TOK/128), 256, 0, stream>>>(
        hh, wbf, P(b,12), xB, xA, L_TOK, CDIM, CMLP, EP_F32RES);
  }
  ln_final_tr<<<L_TOK/8, 256, 0, stream>>>(xA, norm_g, norm_b, (float*)d_out);
}
